// MultiHeadAttention_8306466750797
// MI455X (gfx1250) — matrix operand order repaired and checked by reading, not compiled, not run on hardware
//
#include <hip/hip_runtime.h>
#include <hip/hip_bf16.h>

typedef __attribute__((ext_vector_type(16))) _Float16 v16h;
typedef __attribute__((ext_vector_type(8)))  _Float16 v8h;
typedef __attribute__((ext_vector_type(8)))  float    v8f;
typedef __attribute__((ext_vector_type(4)))  unsigned int u32x4;
typedef __attribute__((ext_vector_type(8)))  int      i32x8;
typedef __attribute__((ext_vector_type(4)))  int      i32x4;

#define NB_     4
#define NT_     4096
#define NHEADS_ 4
#define HDIM_   64
#define HID_    256

__device__ __forceinline__ v8f wmma16(v16h a, v16h b, v8f c) {
  return __builtin_amdgcn_wmma_f32_16x16x32_f16(false, a, false, b, (short)0, c,
                                                false, false);
}

// A-fragment (16x32 f16): lane<16 holds row=lane, K=k0..k0+15;
// lane>=16 holds row=lane-16, K=k0+16..k0+31 (contiguous, matches load_b_f16).
__device__ __forceinline__ v16h load_a_f16(const _Float16* p, int ld, int lane, int k0) {
  int row = lane & 15;
  int kb  = k0 + ((lane & 16) ? 16 : 0);
  const _Float16* r = p + row * ld + kb;
  v8h lo = *(const v8h*)(r);
  v8h hi = *(const v8h*)(r + 8);
  v16h out;
#pragma unroll
  for (int i = 0; i < 8; ++i) { out[i] = lo[i]; out[i + 8] = hi[i]; }
  return out;
}

// B-fragment (32x16 f16) from "B^T" storage bt[N][K]: lane = column, halves = K.
__device__ __forceinline__ v16h load_b_f16(const _Float16* bt, int ld, int lane, int k0) {
  int col = lane & 15;
  int kb  = k0 + ((lane & 16) ? 16 : 0);
  return *(const v16h*)(bt + col * ld + kb);
}

// A-fragment with on-the-fly f32 -> f16 conversion.
__device__ __forceinline__ v16h load_a_f32cvt(const float* p, int ld, int lane, int k0) {
  int row = lane & 15;
  int kb  = k0 + ((lane & 16) ? 16 : 0);
  const float* r = p + row * ld + kb;
  v8f lo = *(const v8f*)(r);
  v8f hi = *(const v8f*)(r + 8);
  v16h out;
#pragma unroll
  for (int i = 0; i < 8; ++i) { out[i] = (_Float16)lo[i]; out[i + 8] = (_Float16)hi[i]; }
  return out;
}

// ---------------- Tensor Data Mover helpers ------------------------------------------
typedef __attribute__((address_space(3))) char lds_char;
__device__ __forceinline__ unsigned lds_off(const void* p) {
  return (unsigned)(unsigned long long)(const lds_char*)p;   // LDS byte offset
}

// Issue a 2D f16 tile DMA: tile is (tile_d1 rows) x (tile_d0 contiguous elements),
// row stride = stride0 elements. Tracked by TENSORcnt. D# layout per ISA ch.8.
__device__ __forceinline__ void tdm_load_2d_f16(unsigned ldsaddr, const _Float16* gsrc,
                                                unsigned tile_d0, unsigned tile_d1,
                                                unsigned long long stride0) {
  unsigned long long ga = (unsigned long long)gsrc;
  u32x4 g0;
  g0[0] = 1u;                                               // count=1, user descriptor
  g0[1] = ldsaddr;                                          // lds_addr (bytes)
  g0[2] = (unsigned)(ga & 0xffffffffu);                     // global_addr[31:0]
  g0[3] = (unsigned)((ga >> 32) & 0x1ffffffu) | (2u << 30); // addr[56:32] | type=2
  i32x8 g1;
  g1[0] = (int)(1u << 16);                                  // wg_mask=0, data_size=2B
  g1[1] = (int)(tile_d0 << 16);                             // tensor_dim0[15:0]
  g1[2] = (int)((tile_d0 >> 16) | (tile_d1 << 16));         // dim0 hi | tensor_dim1 lo
  g1[3] = (int)((tile_d1 >> 16) | (tile_d0 << 16));         // dim1 hi | tile_dim0
  g1[4] = (int)(tile_d1);                                   // tile_dim1 | tile_dim2=0
  g1[5] = (int)(unsigned)(stride0 & 0xffffffffu);           // tensor_dim0_stride lo32
  g1[6] = (int)(unsigned)((stride0 >> 32) & 0xffffu);       // stride0 hi16 | stride1 lo=0
  g1[7] = 0;
  i32x4 z4 = {0, 0, 0, 0};
#if __clang_major__ >= 23
  i32x8 z8 = {0, 0, 0, 0, 0, 0, 0, 0};
  __builtin_amdgcn_tensor_load_to_lds(g0, g1, z4, z4, z8, 0);
#else
  __builtin_amdgcn_tensor_load_to_lds(g0, g1, z4, z4, 0);
#endif
}

// ---------------- weight convert + transpose: Wt[n][k] = (f16) W[k][n] ----------------
__global__ void wcvt_kernel(const float* __restrict__ W, _Float16* __restrict__ Wt) {
  int idx = blockIdx.x * blockDim.x + threadIdx.x;   // 0..65535
  int k = idx >> 8, n = idx & 255;
  Wt[n * HID_ + k] = (_Float16)W[idx];
}

// ---------------- fused QKV projection + bias + ELU + L2 norm -------------------------
__global__ __launch_bounds__(128) void qkv_kernel(
    const float* __restrict__ x,
    const _Float16* __restrict__ Wqt, const _Float16* __restrict__ Wkt,
    const _Float16* __restrict__ Wvt,
    const float* __restrict__ bq, const float* __restrict__ bk,
    const float* __restrict__ bv,
    _Float16* __restrict__ Qn, _Float16* __restrict__ Kn, _Float16* __restrict__ Vt) {
  const int lane = threadIdx.x & 31;
  const int h    = threadIdx.x >> 5;
  const int mtx  = blockIdx.y;                 // 0=Q 1=K 2=V
  const int tok0 = blockIdx.x * 16;

  const _Float16* Wt   = (mtx == 0) ? Wqt : (mtx == 1) ? Wkt : Wvt;
  const float*    bias = (mtx == 0) ? bq  : (mtx == 1) ? bk  : bv;

  v8f acc[4] = {};
  const float* xrow = x + tok0 * HID_;
#pragma unroll
  for (int k0 = 0; k0 < HID_; k0 += 32) {
    v16h a = load_a_f32cvt(xrow, HID_, lane, k0);
#pragma unroll
    for (int g = 0; g < 4; ++g) {
      v16h b = load_b_f16(Wt + (h * 64 + g * 16) * HID_, HID_, lane, k0);
      acc[g] = wmma16(a, b, acc[g]);
    }
  }
#pragma unroll
  for (int g = 0; g < 4; ++g) {
    float bb = bias[h * 64 + g * 16 + (lane & 15)];
#pragma unroll
    for (int r = 0; r < 8; ++r) acc[g][r] += bb;
  }

  const int b_   = tok0 >> 12;
  const int t0   = tok0 & (NT_ - 1);
  const int mofs = (lane & 16) ? 8 : 0;

  if (mtx < 2) {
#pragma unroll
    for (int g = 0; g < 4; ++g)
#pragma unroll
      for (int r = 0; r < 8; ++r) {
        float vv  = acc[g][r];
        acc[g][r] = (vv > 0.0f) ? vv : (__expf(vv) - 1.0f);   // ELU
      }
#pragma unroll
    for (int r = 0; r < 8; ++r) {                              // L2 norm over head dim
      float s = 0.0f;
#pragma unroll
      for (int g = 0; g < 4; ++g) s += acc[g][r] * acc[g][r];
      s += __shfl_xor(s, 1); s += __shfl_xor(s, 2);
      s += __shfl_xor(s, 4); s += __shfl_xor(s, 8);
      float rn = 1.0f / (sqrtf(s) + 1e-6f);
#pragma unroll
      for (int g = 0; g < 4; ++g) acc[g][r] *= rn;
    }
    _Float16* out = (mtx == 0 ? Qn : Kn) + ((b_ * NHEADS_ + h) * NT_ + t0) * HDIM_;
#pragma unroll
    for (int g = 0; g < 4; ++g)
#pragma unroll
      for (int r = 0; r < 8; ++r)
        out[(r + mofs) * HDIM_ + g * 16 + (lane & 15)] = (_Float16)acc[g][r];
  } else {
    _Float16* out = Vt + (b_ * NHEADS_ + h) * HDIM_ * NT_;     // V transposed [d][t]
#pragma unroll
    for (int g = 0; g < 4; ++g)
#pragma unroll
      for (int r = 0; r < 8; ++r)
        out[(g * 16 + (lane & 15)) * NT_ + t0 + r + mofs] = (_Float16)acc[g][r];
  }
}

// ---------------- flash attention with TDM-staged K/V tiles ---------------------------
// 4 waves/block share one (b,h); wave 0 DMA's 64x64 f16 K and V tiles into
// double-buffered LDS via tensor_load_to_lds; all waves compute from LDS.
__global__ __launch_bounds__(128) void attn_kernel(
    const _Float16* __restrict__ Qn, const _Float16* __restrict__ Kn,
    const _Float16* __restrict__ Vt, _Float16* __restrict__ ctx) {
  __shared__ _Float16 kbuf[2][64 * 64];        // [key][d]
  __shared__ _Float16 vbuf[2][64 * 64];        // [d][key]
  __shared__ _Float16 smemP[4][16 * 64];
  const int lane = threadIdx.x & 31;
  const int wv   = threadIdx.x >> 5;
  const int tile = blockIdx.x * 4 + wv;        // 0..4095, 4 consecutive q-tiles, same bh
  const int b_   = tile >> 10;
  const int h    = (tile >> 8) & 3;
  const int q0   = (tile & 255) * 16;

  const _Float16* Qh = Qn + (b_ * NHEADS_ + h) * NT_ * HDIM_;
  const _Float16* Kh = Kn + (b_ * NHEADS_ + h) * NT_ * HDIM_;
  const _Float16* Vh = Vt + (b_ * NHEADS_ + h) * HDIM_ * NT_;

  // prologue: DMA first K/V tiles into buffer 0
  if (wv == 0) {
    tdm_load_2d_f16(lds_off(&kbuf[0][0]), Kh, 64, 64, HDIM_);
    tdm_load_2d_f16(lds_off(&vbuf[0][0]), Vh, 64, 64, NT_);
  }

  v16h aq[2];
#pragma unroll
  for (int kk = 0; kk < 2; ++kk) {             // Q pre-scaled by 1/sqrt(64) (exact)
    v16h a = load_a_f16(Qh + q0 * HDIM_, HDIM_, lane, kk * 32);
#pragma unroll
    for (int i = 0; i < 16; ++i) a[i] = a[i] * (_Float16)0.125f;
    aq[kk] = a;
  }

  v8f cacc[4] = {};
  float mrow[8], lrow[8];
#pragma unroll
  for (int r = 0; r < 8; ++r) { mrow[r] = -1e30f; lrow[r] = 0.0f; }
  _Float16* pP = &smemP[wv][0];

  for (int it = 0; it < NT_ / 64; ++it) {
    const int buf = it & 1;
    if (wv == 0) __builtin_amdgcn_s_wait_tensorcnt(0);  // current buffer DMA done
    __syncthreads();                                    // publish; prev buffer free
    if (wv == 0 && it + 1 < NT_ / 64) {                 // prefetch next tiles
      const int kn = (it + 1) * 64;
      tdm_load_2d_f16(lds_off(&kbuf[buf ^ 1][0]), Kh + kn * HDIM_, 64, 64, HDIM_);
      tdm_load_2d_f16(lds_off(&vbuf[buf ^ 1][0]), Vh + kn, 64, 64, NT_);
    }
    const _Float16* Kt = &kbuf[buf][0];
    const _Float16* Vtile = &vbuf[buf][0];

    v8f s[4];
#pragma unroll
    for (int kt = 0; kt < 4; ++kt) {           // S = (Q/8) @ K^T from LDS
      v8f z = {};
      z = wmma16(aq[0], load_b_f16(Kt + kt * 16 * HDIM_, HDIM_, lane, 0),  z);
      z = wmma16(aq[1], load_b_f16(Kt + kt * 16 * HDIM_, HDIM_, lane, 32), z);
      s[kt] = z;
    }
#pragma unroll
    for (int r = 0; r < 8; ++r) {              // online softmax (f32)
      float mx = fmaxf(fmaxf(s[0][r], s[1][r]), fmaxf(s[2][r], s[3][r]));
      mx = fmaxf(mx, __shfl_xor(mx, 1)); mx = fmaxf(mx, __shfl_xor(mx, 2));
      mx = fmaxf(mx, __shfl_xor(mx, 4)); mx = fmaxf(mx, __shfl_xor(mx, 8));
      float mnew = fmaxf(mrow[r], mx);
      float fs   = __expf(mrow[r] - mnew);
      mrow[r] = mnew;
      float ssum = 0.0f;
#pragma unroll
      for (int g = 0; g < 4; ++g) {
        float p = __expf(s[g][r] - mnew);
        s[g][r] = p;  ssum += p;
      }
      ssum += __shfl_xor(ssum, 1); ssum += __shfl_xor(ssum, 2);
      ssum += __shfl_xor(ssum, 4); ssum += __shfl_xor(ssum, 8);
      lrow[r] = lrow[r] * fs + ssum;
#pragma unroll
      for (int g = 0; g < 4; ++g) cacc[g][r] *= fs;
    }
    // P (C-fragment layout) -> LDS, re-read as A-fragments
#pragma unroll
    for (int g = 0; g < 4; ++g)
#pragma unroll
      for (int r = 0; r < 8; ++r) {
        int m = r + ((lane & 16) ? 8 : 0);
        pP[m * 64 + g * 16 + (lane & 15)] = (_Float16)s[g][r];
      }
    asm volatile("s_wait_dscnt 0x0" ::: "memory");
    v16h ap0 = load_a_f16(pP, 64, lane, 0);
    v16h ap1 = load_a_f16(pP, 64, lane, 32);
#pragma unroll
    for (int g = 0; g < 4; ++g) {              // ctx += P @ V from LDS
      cacc[g] = wmma16(ap0, load_b_f16(Vtile + g * 16 * 64, 64, lane, 0),  cacc[g]);
      cacc[g] = wmma16(ap1, load_b_f16(Vtile + g * 16 * 64, 64, lane, 32), cacc[g]);
    }
  }

#pragma unroll
  for (int r = 0; r < 8; ++r) {
    float inv = 1.0f / lrow[r];
#pragma unroll
    for (int g = 0; g < 4; ++g) cacc[g][r] *= inv;
  }
  _Float16* crow = ctx + (b_ * NT_ + q0) * HID_ + h * HDIM_;
#pragma unroll
  for (int g = 0; g < 4; ++g)
#pragma unroll
    for (int r = 0; r < 8; ++r) {
      int m = r + ((lane & 16) ? 8 : 0);
      crow[m * HID_ + g * 16 + (lane & 15)] = (_Float16)cacc[g][r];
    }
}

// ---------------- out projection + bias + residual + LayerNorm -----------------------
__global__ __launch_bounds__(128) void oproj_ln_kernel(
    const _Float16* __restrict__ ctx, const _Float16* __restrict__ Wot,
    const float* __restrict__ bo, const float* __restrict__ x,
    const float* __restrict__ gamma, const float* __restrict__ beta,
    float* __restrict__ out) {
  __shared__ float sres[16][HID_];
  __shared__ float smu[16], srstd[16];
  const int lane = threadIdx.x & 31;
  const int wv   = threadIdx.x >> 5;
  const int tok0 = blockIdx.x * 16;

  v8f acc[4] = {};
#pragma unroll
  for (int k0 = 0; k0 < HID_; k0 += 32) {
    v16h a = load_a_f16(ctx + tok0 * HID_, HID_, lane, k0);
#pragma unroll
    for (int g = 0; g < 4; ++g) {
      v16h b = load_b_f16(Wot + (wv * 64 + g * 16) * HID_, HID_, lane, k0);
      acc[g] = wmma16(a, b, acc[g]);
    }
  }
#pragma unroll
  for (int g = 0; g < 4; ++g) {
    int c = wv * 64 + g * 16 + (lane & 15);
    float bb = bo[c];
#pragma unroll
    for (int r = 0; r < 8; ++r) {
      int m = r + ((lane & 16) ? 8 : 0);
      sres[m][c] = acc[g][r] + bb + x[(tok0 + m) * HID_ + c];   // residual
    }
  }
  __syncthreads();
  if (threadIdx.x < 16) {
    int m = threadIdx.x;
    float s = 0.0f, sq = 0.0f;
    for (int c = 0; c < HID_; ++c) { float v = sres[m][c]; s += v; sq += v * v; }
    float mu  = s * (1.0f / HID_);
    float var = sq * (1.0f / HID_) - mu * mu;
    smu[m]   = mu;
    srstd[m] = rsqrtf(var + 1e-12f);
  }
  __syncthreads();
#pragma unroll
  for (int i = 0; i < 32; ++i) {
    int idx = threadIdx.x + i * 128;            // 16*256 / 128 threads
    int m = idx >> 8, c = idx & 255;
    out[(tok0 + m) * HID_ + c] = (sres[m][c] - smu[m]) * srstd[m] * gamma[c] + beta[c];
  }
}

extern "C" void kernel_launch(void* const* d_in, const int* in_sizes, int n_in,
                              void* d_out, int out_size, void* d_ws, size_t ws_size,
                              hipStream_t stream) {
  const float* x     = (const float*)d_in[0];
  const float* Wq    = (const float*)d_in[1];
  const float* bq    = (const float*)d_in[2];
  const float* Wk    = (const float*)d_in[3];
  const float* bk    = (const float*)d_in[4];
  const float* Wv    = (const float*)d_in[5];
  const float* bv    = (const float*)d_in[6];
  const float* Wo    = (const float*)d_in[7];
  const float* bo    = (const float*)d_in[8];
  const float* gamma = (const float*)d_in[9];
  const float* beta  = (const float*)d_in[10];

  _Float16* Wqt = (_Float16*)d_ws;                 // 4 x 64K f16 transposed weights
  _Float16* Wkt = Wqt + 65536;
  _Float16* Wvt = Wkt + 65536;
  _Float16* Wot = Wvt + 65536;
  _Float16* Qn  = Wot + 65536;                     // [B,NH,T,64] f16, ELU+norm'd
  _Float16* Kn  = Qn + 4194304;
  _Float16* Vt  = Kn + 4194304;                    // [B,NH,64,T] f16
  _Float16* Ctx = Vt + 4194304;                    // [B,T,256]   f16

  wcvt_kernel<<<256, 256, 0, stream>>>(Wq, Wqt);
  wcvt_kernel<<<256, 256, 0, stream>>>(Wk, Wkt);
  wcvt_kernel<<<256, 256, 0, stream>>>(Wv, Wvt);
  wcvt_kernel<<<256, 256, 0, stream>>>(Wo, Wot);

  qkv_kernel<<<dim3(1024, 3), 128, 0, stream>>>(x, Wqt, Wkt, Wvt, bq, bk, bv, Qn, Kn, Vt);
  attn_kernel<<<1024, 128, 0, stream>>>(Qn, Kn, Vt, Ctx);
  oproj_ln_kernel<<<1024, 128, 0, stream>>>(Ctx, Wot, bo, x, gamma, beta, (float*)d_out);
}
